// SelfAttention_19387482375149
// MI455X (gfx1250) — matrix operand order repaired and checked by reading, not compiled, not run on hardware
//
#include <hip/hip_runtime.h>
#include <hip/hip_bf16.h>

// ---------------------------------------------------------------------------
// Types for CDNA5 WMMA (wave32): A/B = 16 bf16 per lane, C/D = 8 f32 per lane
// ---------------------------------------------------------------------------
typedef __attribute__((ext_vector_type(16))) __bf16 v16bf;
typedef __attribute__((ext_vector_type(8)))  float  v8f;
typedef __attribute__((ext_vector_type(4)))  int    v4i;

union FragAB {
    uint4  u[2];   // 32 bytes
    v16bf  v;
};

#ifndef __has_builtin
#define __has_builtin(x) 0
#endif

#if defined(__HIP_DEVICE_COMPILE__) && __has_builtin(__builtin_amdgcn_global_load_async_to_lds_b128)
#define HAS_ASYNC_LDS 1
#else
#define HAS_ASYNC_LDS 0
#endif

#if HAS_ASYNC_LDS
__device__ __forceinline__ void async_cp16(void* lds_dst, const void* glb_src) {
    __builtin_amdgcn_global_load_async_to_lds_b128(
        (__attribute__((address_space(1))) v4i*)glb_src,
        (__attribute__((address_space(3))) v4i*)lds_dst,
        0, 0);
}
__device__ __forceinline__ void wait_async0() {
#if __has_builtin(__builtin_amdgcn_s_wait_asynccnt)
    __builtin_amdgcn_s_wait_asynccnt(0);
#else
    asm volatile("s_wait_asynccnt 0" ::: "memory");
#endif
}
#else
__device__ __forceinline__ void async_cp16(void*, const void*) {}
__device__ __forceinline__ void wait_async0() {}
#endif

__device__ __forceinline__ __bf16 f2bf(float f) {
    union { float f; unsigned int u; } c; c.f = f;
    unsigned int r = c.u + 0x7FFFu + ((c.u >> 16) & 1u);  // round-to-nearest-even
    union { unsigned short s; __bf16 b; } o; o.s = (unsigned short)(r >> 16);
    return o.b;
}

__device__ __forceinline__ float bf2f(__bf16 b) {
    union { unsigned short s; __bf16 b; } i; i.b = b;
    union { unsigned int u; float f; } o; o.u = ((unsigned int)i.s) << 16;
    return o.f;
}

// ---------------------------------------------------------------------------
// Elementwise fp32 -> bf16 conversion
// ---------------------------------------------------------------------------
__global__ void f32_to_bf16_kernel(const float* __restrict__ in,
                                   __bf16* __restrict__ out, int n) {
    int i = blockIdx.x * blockDim.x + threadIdx.x;
    int stride = gridDim.x * blockDim.x;
    for (; i < n; i += stride) out[i] = f2bf(in[i]);
}

// ---------------------------------------------------------------------------
// WMMA GEMM:  C[m,n] = scale * sum_k A[m,k] * Bop[k,n]  (+ bias[n])
//   BT=true :  B given as [N,K] row-major (torch-Linear weight / K-matrix)
//   BT=false:  B given as [K,N] row-major (V-matrix) -> transposed LDS staging
// Block tile 256x128, 8 waves (4Mx2N), wave tile 64x64 (4x4 WMMA 16x16 tiles).
// Double-buffered LDS staged with GLOBAL_LOAD_ASYNC_TO_LDS_B128 (ASYNCcnt),
// one barrier per 32-K chunk; zero staging VGPRs on the async path.
// M multiple of 256; N multiple of 128; K multiple of 32.
// ---------------------------------------------------------------------------
template <bool BT, bool HAS_BIAS, bool OUT_F32>
__global__ __launch_bounds__(256, 1)
void gemm_wmma_bf16_kernel(const __bf16* __restrict__ Ag,
                           const __bf16* __restrict__ Bg,
                           const float* __restrict__ bias,
                           float scale,
                           void* __restrict__ Cout,
                           int M, int N, int K,
                           long long bsA, long long bsB, long long bsC) {
    __shared__ alignas(16) __bf16 Atile[2][256 * 32];   // 2 x 16 KiB
    __shared__ alignas(16) __bf16 Btile[2][128 * 32];   // 2 x  8 KiB

    const int tid  = threadIdx.x;
    const int lane = tid & 31;
    const int wave = tid >> 5;
    const int wm   = wave >> 1;   // 0..3 -> M offset wm*64
    const int wn   = wave & 1;    // 0..1 -> N offset wn*64
    const int bm   = blockIdx.y * 256;
    const int bn   = blockIdx.x * 128;
    const int z    = blockIdx.z;

    const __bf16* __restrict__ A = Ag + (long long)z * bsA;
    const __bf16* __restrict__ B = Bg + (long long)z * bsB;

    v8f acc[4][4];
#pragma unroll
    for (int i = 0; i < 4; i++)
#pragma unroll
        for (int j = 0; j < 4; j++)
            acc[i][j] = v8f{0.f, 0.f, 0.f, 0.f, 0.f, 0.f, 0.f, 0.f};

    const int lm  = lane & 15;    // row/col within 16x16 tile
    const int lhi = lane >> 4;    // 0 or 1 (K-half select)

    // staging coordinates (32-bit element offsets; all buffers < 2^31 elems)
    const int acol = (tid & 3) * 8;      // A/BT: 4 threads span 32 k
    const int arow = tid >> 2;           // A: rows 0..63 (+p*64); BT: 2 passes
    const int tnc  = (tid & 15) * 8;     // B (non-BT): 8 n per thread
    const int tk0  = tid >> 4;           // B (non-BT): k rows 0..15 (+p*16)

    const unsigned aoff  = (unsigned)(bm + arow) * (unsigned)K + acol;  // + p*64*K + kc
    const unsigned boffT = (unsigned)(bn + arow) * (unsigned)K + acol;  // BT path
    const unsigned boffN = (unsigned)tk0 * (unsigned)N + bn + tnc;      // non-BT

    // ---- prologue: stage chunk 0 into buffer 0 ----
    if constexpr (HAS_ASYNC_LDS != 0) {
#pragma unroll
        for (int p = 0; p < 4; p++)
            async_cp16(&Atile[0][(arow + p * 64) * 32 + acol], &A[aoff + p * 64 * K]);
        if constexpr (BT) {
#pragma unroll
            for (int p = 0; p < 2; p++)
                async_cp16(&Btile[0][(arow + p * 64) * 32 + acol], &B[boffT + p * 64 * K]);
        }
    } else {
#pragma unroll
        for (int p = 0; p < 4; p++)
            *(uint4*)&Atile[0][(arow + p * 64) * 32 + acol] =
                *(const uint4*)&A[aoff + p * 64 * K];
        if constexpr (BT) {
#pragma unroll
            for (int p = 0; p < 2; p++)
                *(uint4*)&Btile[0][(arow + p * 64) * 32 + acol] =
                    *(const uint4*)&B[boffT + p * 64 * K];
        }
    }
    if constexpr (!BT) {
#pragma unroll
        for (int p = 0; p < 2; p++) {
            const int k = tk0 + p * 16;
            uint4 v = *(const uint4*)&B[boffN + p * 16 * N];
            const __bf16* e = (const __bf16*)&v;
#pragma unroll
            for (int j = 0; j < 8; j++) Btile[0][(tnc + j) * 32 + k] = e[j];
        }
    }
    wait_async0();
    __syncthreads();

    int buf = 0;
    for (int kc = 0; kc < K; kc += 32) {
        const bool has_next = (kc + 32) < K;
        const int  kn = kc + 32;
        const int  nb = buf ^ 1;

        // ---- issue staging of the next chunk into buf^1 ----
        uint4 rb[2];
        if (has_next) {
            if constexpr (HAS_ASYNC_LDS != 0) {
#pragma unroll
                for (int p = 0; p < 4; p++)
                    async_cp16(&Atile[nb][(arow + p * 64) * 32 + acol],
                               &A[aoff + kn + p * 64 * K]);
                if constexpr (BT) {
#pragma unroll
                    for (int p = 0; p < 2; p++)
                        async_cp16(&Btile[nb][(arow + p * 64) * 32 + acol],
                                   &B[boffT + kn + p * 64 * K]);
                }
            }
            if constexpr (!BT) {   // transposed staging: must go through VGPRs
#pragma unroll
                for (int p = 0; p < 2; p++)
                    rb[p] = *(const uint4*)&B[boffN + (unsigned)(kn + p * 16) * N];
            }
        }

        // ---- fragments + 16 WMMAs from buffer `buf` ----
        const __bf16* At = &Atile[buf][0];
        const __bf16* Bt = &Btile[buf][0];

        FragAB bfr[4];
#pragma unroll
        for (int nsub = 0; nsub < 4; nsub++) {
            const int nrow = wn * 64 + nsub * 16 + lm;
            const int ko   = lhi * 8;
            bfr[nsub].u[0] = *(const uint4*)&Bt[nrow * 32 + ko];
            bfr[nsub].u[1] = *(const uint4*)&Bt[nrow * 32 + ko + 16];
        }
#pragma unroll
        for (int msub = 0; msub < 4; msub++) {
            FragAB afr;
            const int mrow = wm * 64 + msub * 16 + lm;
            const int ko   = lhi * 8;
            afr.u[0] = *(const uint4*)&At[mrow * 32 + ko];       // VGPR0-3: K=ko..ko+7
            afr.u[1] = *(const uint4*)&At[mrow * 32 + ko + 16];  // VGPR4-7: K=16+ko..
#pragma unroll
            for (int nsub = 0; nsub < 4; nsub++) {
                acc[msub][nsub] = __builtin_amdgcn_wmma_f32_16x16x32_bf16(
                    false, afr.v, false, bfr[nsub].v,
                    (short)0, acc[msub][nsub], false, false);
            }
        }

        // ---- commits + synchronization ----
        if (has_next) {
            if constexpr (HAS_ASYNC_LDS == 0) {
                if constexpr (BT) {
                    // fallback: re-load + store (no async support)
#pragma unroll
                    for (int p = 0; p < 4; p++)
                        *(uint4*)&Atile[nb][(arow + p * 64) * 32 + acol] =
                            *(const uint4*)&A[aoff + kn + p * 64 * K];
#pragma unroll
                    for (int p = 0; p < 2; p++)
                        *(uint4*)&Btile[nb][(arow + p * 64) * 32 + acol] =
                            *(const uint4*)&B[boffT + kn + p * 64 * K];
                } else {
#pragma unroll
                    for (int p = 0; p < 4; p++)
                        *(uint4*)&Atile[nb][(arow + p * 64) * 32 + acol] =
                            *(const uint4*)&A[aoff + kn + p * 64 * K];
                }
            }
            if constexpr (!BT) {
#pragma unroll
                for (int p = 0; p < 2; p++) {
                    const int k = tk0 + p * 16;
                    const __bf16* e = (const __bf16*)&rb[p];
#pragma unroll
                    for (int j = 0; j < 8; j++) Btile[nb][(tnc + j) * 32 + k] = e[j];
                }
            }
            wait_async0();
            __syncthreads();
        }
        buf ^= 1;
    }

    // ---- store: C/D layout -> lane(0-15)=N col, VGPR r -> M = r + 8*(lane>=16)
#pragma unroll
    for (int msub = 0; msub < 4; msub++) {
#pragma unroll
        for (int nsub = 0; nsub < 4; nsub++) {
            const int ncol  = bn + wn * 64 + nsub * 16 + lm;
            const int mbase = bm + wm * 64 + msub * 16 + 8 * lhi;
            float bv = 0.f;
            if constexpr (HAS_BIAS) bv = bias[ncol];
#pragma unroll
            for (int r = 0; r < 8; r++) {
                const float val = acc[msub][nsub][r] * scale + bv;
                const size_t idx = (size_t)(mbase + r) * N + ncol + (size_t)z * bsC;
                if constexpr (OUT_F32) ((float*)Cout)[idx]  = val;
                else                   ((__bf16*)Cout)[idx] = f2bf(val);
            }
        }
    }
}

// ---------------------------------------------------------------------------
// In-place row softmax over bf16 scores (row length 2048, 256 thr, 8 ele/thr)
// ---------------------------------------------------------------------------
__global__ __launch_bounds__(256)
void softmax_rows_kernel(__bf16* __restrict__ s) {
    __shared__ float red[256];
    const int t = threadIdx.x;
    __bf16* p = s + (size_t)blockIdx.x * 2048;

    float v[8];
    float m = -3.4e38f;
#pragma unroll
    for (int j = 0; j < 8; j++) { v[j] = bf2f(p[t + j * 256]); m = fmaxf(m, v[j]); }
    red[t] = m; __syncthreads();
    for (int off = 128; off > 0; off >>= 1) {
        if (t < off) red[t] = fmaxf(red[t], red[t + off]);
        __syncthreads();
    }
    const float rowmax = red[0];
    __syncthreads();

    float sum = 0.f;
#pragma unroll
    for (int j = 0; j < 8; j++) { v[j] = __expf(v[j] - rowmax); sum += v[j]; }
    red[t] = sum; __syncthreads();
    for (int off = 128; off > 0; off >>= 1) {
        if (t < off) red[t] += red[t + off];
        __syncthreads();
    }
    const float inv = 1.f / red[0];
#pragma unroll
    for (int j = 0; j < 8; j++) p[t + j * 256] = f2bf(v[j] * inv);
}

// ---------------------------------------------------------------------------
// Launcher: x,wq,bq,wk,bk,wv,bv,wo,bo -> out [4,2048,1024] f32
// ---------------------------------------------------------------------------
extern "C" void kernel_launch(void* const* d_in, const int* in_sizes, int n_in,
                              void* d_out, int out_size, void* d_ws, size_t ws_size,
                              hipStream_t stream) {
    constexpr int Bz = 4, S = 2048, D = 1024;
    constexpr int MS = Bz * S;                // 8192 flattened rows for projections
    const float* x    = (const float*)d_in[0];
    const float* w_q  = (const float*)d_in[1];
    const float* b_q  = (const float*)d_in[2];
    const float* w_k  = (const float*)d_in[3];
    const float* b_k  = (const float*)d_in[4];
    const float* w_v  = (const float*)d_in[5];
    const float* b_v  = (const float*)d_in[6];
    const float* w_o  = (const float*)d_in[7];
    const float* b_o  = (const float*)d_in[8];

    const size_t MB = 1ull << 20;
    char* ws = (char*)d_ws;
    __bf16* xb  = (__bf16*)(ws);             // 16 MiB : x (bf16), later reused as attn-out
    __bf16* wqb = (__bf16*)(ws + 16 * MB);   //  2 MiB
    __bf16* wkb = (__bf16*)(ws + 18 * MB);
    __bf16* wvb = (__bf16*)(ws + 20 * MB);
    __bf16* wob = (__bf16*)(ws + 22 * MB);
    __bf16* Qb  = (__bf16*)(ws + 24 * MB);   // 16 MiB
    __bf16* Kb  = (__bf16*)(ws + 40 * MB);   // 16 MiB
    __bf16* Vb  = (__bf16*)(ws + 56 * MB);   // 16 MiB
    __bf16* Sb  = (__bf16*)(ws + 72 * MB);   // 32 MiB (scores -> attn in place)
    __bf16* Ob  = xb;                        // attn@V reuses xb region

    // 1) fp32 -> bf16 conversions
    f32_to_bf16_kernel<<<4096, 256, 0, stream>>>(x,   xb,  Bz * S * D);
    f32_to_bf16_kernel<<<1024, 256, 0, stream>>>(w_q, wqb, D * D);
    f32_to_bf16_kernel<<<1024, 256, 0, stream>>>(w_k, wkb, D * D);
    f32_to_bf16_kernel<<<1024, 256, 0, stream>>>(w_v, wvb, D * D);
    f32_to_bf16_kernel<<<1024, 256, 0, stream>>>(w_o, wob, D * D);

    // 2) Q/K/V projections: [8192,1024] = xb @ W^T + b   (bf16 out)
    dim3 gProj(D / 128, MS / 256, 1);
    gemm_wmma_bf16_kernel<true, true, false><<<gProj, 256, 0, stream>>>(
        xb, wqb, b_q, 1.0f, Qb, MS, D, D, 0, 0, 0);
    gemm_wmma_bf16_kernel<true, true, false><<<gProj, 256, 0, stream>>>(
        xb, wkb, b_k, 1.0f, Kb, MS, D, D, 0, 0, 0);
    gemm_wmma_bf16_kernel<true, true, false><<<gProj, 256, 0, stream>>>(
        xb, wvb, b_v, 1.0f, Vb, MS, D, D, 0, 0, 0);

    // 3) scores = (Q @ K^T) / sqrt(D)   per batch (bf16 out, 32 MiB, L2-resident)
    dim3 gScore(S / 128, S / 256, Bz);
    gemm_wmma_bf16_kernel<true, false, false><<<gScore, 256, 0, stream>>>(
        Qb, Kb, nullptr, 0.03125f, Sb, S, S, D,
        (long long)S * D, (long long)S * D, (long long)S * S);

    // 4) softmax rows in place
    softmax_rows_kernel<<<Bz * S, 256, 0, stream>>>(Sb);

    // 5) attn @ V  (B is [K,N] row-major -> transposed staging path)
    dim3 gPV(D / 128, S / 256, Bz);
    gemm_wmma_bf16_kernel<false, false, false><<<gPV, 256, 0, stream>>>(
        Sb, Vb, nullptr, 1.0f, Ob, S, D, S,
        (long long)S * S, (long long)S * D, (long long)S * D);

    // 6) out projection: f32 output + bias to d_out
    gemm_wmma_bf16_kernel<true, true, true><<<gProj, 256, 0, stream>>>(
        Ob, wob, b_o, 1.0f, d_out, MS, D, D, 0, 0, 0);
}
